// WindowAttention1D_86242943303937
// MI455X (gfx1250) — matrix operand order repaired and checked by reading, not compiled, not run on hardware
//
#include <hip/hip_runtime.h>

typedef __bf16 bf16x16 __attribute__((ext_vector_type(16)));
typedef __bf16 bf16x2  __attribute__((ext_vector_type(2)));
typedef float  f32x8   __attribute__((ext_vector_type(8)));
typedef float  f32x2   __attribute__((ext_vector_type(2)));

__device__ __forceinline__ unsigned short f2bf(float f) {
  union { __bf16 h; unsigned short s; } u;
  u.h = (__bf16)f;                       // native cvt, RTE
  return u.s;
}
__device__ __forceinline__ unsigned pk2(float a, float b) {
  union { bf16x2 v; unsigned u; } c;
  f32x2 t = {a, b};
  c.v = __builtin_convertvector(t, bf16x2);  // v_cvt_pk_bf16_f32 if available
  return c.u;
}

union Frag {
  int4 q[2];
  bf16x16 v;
};

struct __align__(16) Smem {
  unsigned short qs[64 * 264];     // q (scaled) bf16  [token][chan]
  unsigned short ks[64 * 264];     // k bf16          [token][chan]
  unsigned short vsT[256 * 72];    // v bf16 transposed [chan][token]
  unsigned short ps[8 * 64 * 72];  // softmax probs bf16 [head][query][key]
  unsigned short xo[64 * 264];     // phase A/B: x bf16 ; phase D/E: attn-out bf16
  float rpb[127 * 8];              // relative position bias table
};

// A-matrix fragment (16x32 bf16), row-major source with contiguous K.
// Layout: lanes<16: V0-7 K=0..15 ; lanes>=16: K=16..31 (matches ldB).
__device__ __forceinline__ bf16x16 ldA(const unsigned short* base, int row,
                                       int stride, int k0, int lane) {
  Frag f;
  const unsigned short* p = base + row * stride + k0 + ((lane & 16) ? 16 : 0);
  f.q[0] = *(const int4*)(p);
  f.q[1] = *(const int4*)(p + 8);
  return f.v;
}
// B-matrix fragment (32x16 bf16), source stored per-column with contiguous K.
// Layout (matches ldA, not ISA table order): lane=col, lanes<16: V0-7 K=0..15 ; lanes>=16: K=16..31.
__device__ __forceinline__ bf16x16 ldB(const unsigned short* base, int col,
                                       int stride, int k0, int lane) {
  Frag f;
  const unsigned short* p = base + col * stride + k0 + ((lane & 16) ? 16 : 0);
  f.q[0] = *(const int4*)(p);
  f.q[1] = *(const int4*)(p + 8);
  return f.v;
}
// Fallback: B-matrix fragment from row-major f32 global weights w[K][n].
__device__ __forceinline__ bf16x16 ldBw(const float* w, int ldw, int k0,
                                        int n0, int lane) {
  Frag f;
  const float* p = w + (size_t)(k0 + ((lane & 16) ? 16 : 0)) * ldw + n0 + (lane & 15);
  unsigned u[8];
#pragma unroll
  for (int v = 0; v < 8; ++v)
    u[v] = pk2(p[(2 * v) * ldw], p[(2 * v + 1) * ldw]);
  f.q[0] = make_int4((int)u[0], (int)u[1], (int)u[2], (int)u[3]);
  f.q[1] = make_int4((int)u[4], (int)u[5], (int)u[6], (int)u[7]);
  return f.v;
}

__device__ __forceinline__ f32x8 wmma_bf16(bf16x16 a, bf16x16 b, f32x8 c) {
  return __builtin_amdgcn_wmma_f32_16x16x32_bf16(false, a, false, b, (short)0,
                                                 c, false, false);
}

// ---- prep: convert + transpose weights to bf16 in workspace ----
// wqT[n][k] (n<768,k<256) then wpT[n][k] (n<256,k<256), K contiguous.
__global__ void __launch_bounds__(256)
prep_weights(const float* __restrict__ qkv_w, const float* __restrict__ proj_w,
             unsigned short* __restrict__ ws) {
  int idx = blockIdx.x * 256 + threadIdx.x;  // 0 .. 262143
  if (idx < 768 * 256) {
    int n = idx >> 8, k = idx & 255;
    ws[idx] = f2bf(qkv_w[k * 768 + n]);
  } else {
    int j = idx - 768 * 256;
    int n = j >> 8, k = j & 255;
    ws[768 * 256 + j] = f2bf(proj_w[k * 256 + n]);
  }
}

template <bool WS>
__global__ void __launch_bounds__(256)
win_attn_kernel(const float* __restrict__ x, const float* __restrict__ qkv_w,
                const float* __restrict__ qkv_b, const float* __restrict__ proj_w,
                const float* __restrict__ proj_b, const float* __restrict__ rpb_table,
                float* __restrict__ out, const unsigned short* __restrict__ wqT,
                const unsigned short* __restrict__ wpT) {
  __shared__ Smem sm;
  const int tid  = threadIdx.x;
  const int wave = __builtin_amdgcn_readfirstlane(tid >> 5);  // scalar
  const int lane = tid & 31;
  const int ln   = lane & 15;
  const int off8 = (lane & 16) ? 8 : 0;
  const int b    = blockIdx.x;
  const float scale = 0.17677669529663687f;  // 1/sqrt(32)
  const f32x8 zro = {0, 0, 0, 0, 0, 0, 0, 0};

  // ---------- Phase A: stage x (f32 -> bf16) and rpb into LDS ----------
  const float* xb = x + (size_t)b * (64 * 256);
#pragma unroll
  for (int j = 0; j < 16; ++j) {
    int idx = tid + 256 * j;              // 4096 float4 chunks
    int row = idx >> 6, c4 = idx & 63;
    float4 v = *(const float4*)(xb + idx * 4);
    uint2 d;
    d.x = pk2(v.x, v.y);
    d.y = pk2(v.z, v.w);
    *(uint2*)(sm.xo + row * 264 + c4 * 4) = d;
  }
  for (int idx = tid; idx < 127 * 8; idx += 256) sm.rpb[idx] = rpb_table[idx];
  __syncthreads();

  // ---------- Phase B: QKV GEMM  (64x256)@(256x768) + bias ----------
#pragma unroll 1
  for (int i = 0; i < 6; ++i) {
    int g0 = (wave + 8 * i) * 16;         // output column tile base in [0,768)
    f32x8 acc[4] = {zro, zro, zro, zro};
    for (int kk = 0; kk < 8; ++kk) {
      bf16x16 bfr;
      if (WS) bfr = ldB(wqT, g0 + ln, 256, kk * 32, lane);
      else    bfr = ldBw(qkv_w, 768, kk * 32, g0, lane);
#pragma unroll
      for (int mt = 0; mt < 4; ++mt) {
        bf16x16 afr = ldA(sm.xo, mt * 16 + ln, 264, kk * 32, lane);
        acc[mt] = wmma_bf16(afr, bfr, acc[mt]);
      }
    }
    int g = g0 + ln;
    float bias = qkv_b[g];
    if (g0 < 256) {                        // Q: scale, store row-major
#pragma unroll
      for (int mt = 0; mt < 4; ++mt)
#pragma unroll
        for (int r = 0; r < 8; ++r) {
          int m = mt * 16 + r + off8;
          sm.qs[m * 264 + g] = f2bf((acc[mt][r] + bias) * scale);
        }
    } else if (g0 < 512) {                 // K: store row-major
      int gc = g - 256;
#pragma unroll
      for (int mt = 0; mt < 4; ++mt)
#pragma unroll
        for (int r = 0; r < 8; ++r) {
          int m = mt * 16 + r + off8;
          sm.ks[m * 264 + gc] = f2bf(acc[mt][r] + bias);
        }
    } else {                               // V: store transposed [chan][token]
      int jc = g - 512;
      unsigned short* vrow = sm.vsT + jc * 72;
#pragma unroll
      for (int mt = 0; mt < 4; ++mt) {
        int m0 = mt * 16 + off8;
#pragma unroll
        for (int pi = 0; pi < 4; ++pi)
          *(unsigned*)(vrow + m0 + 2 * pi) =
              pk2(acc[mt][2 * pi] + bias, acc[mt][2 * pi + 1] + bias);
      }
    }
  }
  __syncthreads();

  // ---------- Phase C: scores = qkT + bias, softmax (one head per wave) ----------
  const int h = wave;
  unsigned short* psh = sm.ps + h * (64 * 72);
#pragma unroll 1
  for (int mt = 0; mt < 4; ++mt) {
    bf16x16 afr = ldA(sm.qs, mt * 16 + ln, 264, h * 32, lane);
    f32x8 acc[4];
#pragma unroll
    for (int nt = 0; nt < 4; ++nt) {
      bf16x16 bfr = ldB(sm.ks, nt * 16 + ln, 264, h * 32, lane);
      acc[nt] = wmma_bf16(afr, bfr, zro);
    }
    // relative position bias: row m = query, col n = key
#pragma unroll
    for (int nt = 0; nt < 4; ++nt) {
      int n = nt * 16 + ln;
#pragma unroll
      for (int r = 0; r < 8; ++r) {
        int m = mt * 16 + r + off8;
        acc[nt][r] += sm.rpb[(m - n + 63) * 8 + h];
      }
    }
    // row softmax: reduce across 4 col-tiles + 16-lane group
    float mx[8], sum[8];
#pragma unroll
    for (int r = 0; r < 8; ++r) {
      float v = fmaxf(fmaxf(acc[0][r], acc[1][r]), fmaxf(acc[2][r], acc[3][r]));
      v = fmaxf(v, __shfl_xor(v, 1, 32));
      v = fmaxf(v, __shfl_xor(v, 2, 32));
      v = fmaxf(v, __shfl_xor(v, 4, 32));
      v = fmaxf(v, __shfl_xor(v, 8, 32));
      mx[r] = v;
    }
#pragma unroll
    for (int nt = 0; nt < 4; ++nt)
#pragma unroll
      for (int r = 0; r < 8; ++r) acc[nt][r] = __expf(acc[nt][r] - mx[r]);
#pragma unroll
    for (int r = 0; r < 8; ++r) {
      float s = acc[0][r] + acc[1][r] + acc[2][r] + acc[3][r];
      s += __shfl_xor(s, 1, 32);
      s += __shfl_xor(s, 2, 32);
      s += __shfl_xor(s, 4, 32);
      s += __shfl_xor(s, 8, 32);
      sum[r] = 1.0f / s;
    }
#pragma unroll
    for (int nt = 0; nt < 4; ++nt) {
      int n = nt * 16 + ln;
#pragma unroll
      for (int r = 0; r < 8; ++r) {
        int m = mt * 16 + r + off8;
        psh[m * 72 + n] = f2bf(acc[nt][r] * sum[r]);
      }
    }
  }

  // ---------- Phase D: out_h = P @ v_h  (K = 64 tokens) ----------
#pragma unroll 1
  for (int mt = 0; mt < 4; ++mt) {
#pragma unroll
    for (int ft = 0; ft < 2; ++ft) {
      f32x8 acc = zro;
#pragma unroll
      for (int kt = 0; kt < 2; ++kt) {
        bf16x16 afr = ldA(psh, mt * 16 + ln, 72, kt * 32, lane);
        bf16x16 bfr = ldB(sm.vsT, h * 32 + ft * 16 + ln, 72, kt * 32, lane);
        acc = wmma_bf16(afr, bfr, acc);
      }
      int g = h * 32 + ft * 16 + ln;
#pragma unroll
      for (int r = 0; r < 8; ++r) {
        int m = mt * 16 + r + off8;
        sm.xo[m * 264 + g] = f2bf(acc[r]);
      }
    }
  }
  __syncthreads();

  // ---------- Phase E: proj GEMM (64x256)@(256x256) + bias -> global ----------
  const size_t ob = (size_t)b * (64 * 256);
#pragma unroll 1
  for (int i = 0; i < 2; ++i) {
    int g0 = (wave + 8 * i) * 16;
    f32x8 acc[4] = {zro, zro, zro, zro};
    for (int kk = 0; kk < 8; ++kk) {
      bf16x16 bfr;
      if (WS) bfr = ldB(wpT, g0 + ln, 256, kk * 32, lane);
      else    bfr = ldBw(proj_w, 256, kk * 32, g0, lane);
#pragma unroll
      for (int mt = 0; mt < 4; ++mt) {
        bf16x16 afr = ldA(sm.xo, mt * 16 + ln, 264, kk * 32, lane);
        acc[mt] = wmma_bf16(afr, bfr, acc[mt]);
      }
    }
    int g = g0 + ln;
    float pb = proj_b[g];
#pragma unroll
    for (int mt = 0; mt < 4; ++mt)
#pragma unroll
      for (int r = 0; r < 8; ++r) {
        int m = mt * 16 + r + off8;
        out[ob + m * 256 + g] = acc[mt][r] + pb;
      }
  }
}

extern "C" void kernel_launch(void* const* d_in, const int* in_sizes, int n_in,
                              void* d_out, int out_size, void* d_ws, size_t ws_size,
                              hipStream_t stream) {
  (void)in_sizes; (void)n_in; (void)out_size;
  const float* x      = (const float*)d_in[0];
  const float* qkv_w  = (const float*)d_in[1];
  const float* qkv_b  = (const float*)d_in[2];
  const float* proj_w = (const float*)d_in[3];
  const float* proj_b = (const float*)d_in[4];
  const float* rpb    = (const float*)d_in[5];
  float* out = (float*)d_out;

  const size_t need = (size_t)(768 * 256 + 256 * 256) * 2;  // 512 KB bf16 weights
  if (d_ws && ws_size >= need) {
    unsigned short* wsT = (unsigned short*)d_ws;
    prep_weights<<<dim3(1024), dim3(256), 0, stream>>>(qkv_w, proj_w, wsT);
    win_attn_kernel<true><<<dim3(4096), dim3(256), 0, stream>>>(
        x, qkv_w, qkv_b, proj_w, proj_b, rpb, out, wsT, wsT + 768 * 256);
  } else {
    win_attn_kernel<false><<<dim3(4096), dim3(256), 0, stream>>>(
        x, qkv_w, qkv_b, proj_w, proj_b, rpb, out, nullptr, nullptr);
  }
}
